// Decoder_43336220016932
// MI455X (gfx1250) — matrix operand order repaired and checked by reading, not compiled, not run on hardware
//
#include <hip/hip_runtime.h>
#include <hip/hip_bf16.h>
#include <math.h>

typedef __bf16 bf16;
typedef __attribute__((ext_vector_type(16))) __bf16 v16bf;
typedef __attribute__((ext_vector_type(8)))  float  v8f;

#define B_ 32
#define S_ 64
#define L_ 64
#define U_ 512
#define E_ 256
#define V_ 32000
#define T_ 63

// ---- WMMA 16-bit operand element maps (per CDNA5 ISA 7.12.2) ----
// A-matrix 16x32 bf16: lane half 0 holds K {0..7, 16..23}, half 1 holds K {8..15, 24..31},
// two K per VGPR (lo/hi). j = element index 0..15 within a lane's v16bf.
__device__ __forceinline__ int a_k_of(int lane, int j) {
  int v = j >> 1, hh = j & 1;
  int kk = (v < 4 ? 2 * v : 16 + 2 * (v - 4)) + hh + ((lane >= 16) ? 8 : 0);
  return kk;
}
// B-matrix 32x16 bf16: same K map as A (half 0: K {0..7,16..23}, half 1: K {8..15,24..31}), col = lane%16.
__device__ __forceinline__ int b_k_of(int lane, int j) {
  return a_k_of(lane, j);
}

// ---- Pack f32 row-major (K x N) weight into bf16 WMMA-B tiles: [nb][kb] x (32 lanes x 16 elems)
__global__ __launch_bounds__(256) void pack_b_kernel(const float* __restrict__ W,
                                                     bf16* __restrict__ dst, int K, int N) {
  int numKb = K >> 5, numNb = N >> 4;
  long total = (long)numKb * numNb * 32;
  long lu = (long)blockIdx.x * 256 + threadIdx.x;
  if (lu >= total) return;
  int lane = (int)(lu & 31);
  long tile = lu >> 5;
  int kb = (int)(tile % numKb);
  int nb = (int)(tile / numKb);
  int n = nb * 16 + (lane & 15);
  bf16* o = dst + ((long)nb * numKb + kb) * 512 + lane * 16;
#pragma unroll
  for (int j = 0; j < 16; ++j) {
    int k = kb * 32 + b_k_of(lane, j);
    o[j] = (bf16)W[(long)k * N + n];
  }
}

// ---- Pack f32 row-major (M x K) activations into bf16 WMMA-A tiles: [mb][kb] layout
__global__ __launch_bounds__(256) void pack_a_kernel(const float* __restrict__ A,
                                                     bf16* __restrict__ dst, int M, int K) {
  int numKb = K >> 5, numMb = M >> 4;
  long total = (long)numKb * numMb * 32;
  long lu = (long)blockIdx.x * 256 + threadIdx.x;
  if (lu >= total) return;
  int lane = (int)(lu & 31);
  long tile = lu >> 5;
  int kb = (int)(tile % numKb);
  int mb = (int)(tile / numKb);
  int m = mb * 16 + (lane & 15);
  bf16* o = dst + ((long)mb * numKb + kb) * 512 + lane * 16;
#pragma unroll
  for (int j = 0; j < 16; ++j) {
    int k = kb * 32 + a_k_of(lane, j);
    o[j] = (bf16)A[(long)m * K + k];
  }
}

// ---- Generic bf16 WMMA GEMM: out(M x N) = A(M x K) * B(K x N) [+bias][tanh]
// grid.x = N/256, grid.y = M/32. 256 threads = 8 waves; wave owns 2 n-tiles x 2 m-tiles.
__global__ __launch_bounds__(256) void gemm_bf16_wmma(const bf16* __restrict__ Apk,
                                                      const bf16* __restrict__ Bpk,
                                                      float* __restrict__ out,
                                                      const float* __restrict__ bias,
                                                      int K, long out_stride, int do_tanh) {
  int numKb = K >> 5;
  __shared__ uint4 sA[4096];  // up to 64KB: A block = 2 m-tiles x numKb k-tiles
  // Stage A block into LDS, re-tiled [kb][mb_local] for the inner loop.
  const uint4* src = (const uint4*)(Apk + (long)(blockIdx.y * 2) * numKb * 512);
  int nchunk = 2 * numKb * 64;
  for (int i = threadIdx.x; i < nchunk; i += 256) {
    int tile = i >> 6;          // [mb][kb] source order
    int within = i & 63;
    int mbl = tile / numKb;
    int kb = tile % numKb;
    sA[(kb * 2 + mbl) * 64 + within] = src[i];
  }
  __syncthreads();

  int wave = threadIdx.x >> 5;
  int lane = threadIdx.x & 31;
  long nt0 = (long)blockIdx.x * 16 + wave * 2;
  const bf16* b0p = Bpk + nt0 * numKb * 512 + lane * 16;
  const bf16* b1p = b0p + (long)numKb * 512;
  const bf16* a0p = (const bf16*)sA + lane * 16;

  v8f acc00 = {}, acc01 = {}, acc10 = {}, acc11 = {};
  for (int kb = 0; kb < numKb; ++kb) {
    v16bf a0 = *(const v16bf*)(a0p + (kb * 2 + 0) * 512);
    v16bf a1 = *(const v16bf*)(a0p + (kb * 2 + 1) * 512);
    v16bf b0 = *(const v16bf*)(b0p + kb * 512);
    v16bf b1 = *(const v16bf*)(b1p + kb * 512);
    acc00 = __builtin_amdgcn_wmma_f32_16x16x32_bf16(false, a0, false, b0, (short)0, acc00, false, false);
    acc01 = __builtin_amdgcn_wmma_f32_16x16x32_bf16(false, a0, false, b1, (short)0, acc01, false, false);
    acc10 = __builtin_amdgcn_wmma_f32_16x16x32_bf16(false, a1, false, b0, (short)0, acc10, false, false);
    acc11 = __builtin_amdgcn_wmma_f32_16x16x32_bf16(false, a1, false, b1, (short)0, acc11, false, false);
  }
  // Epilogue: C layout — lane%16 = column, VGPR r = row (+8 for lane>=16)
  int ncol0 = (int)(nt0 * 16) + (lane & 15);
  int ncol1 = ncol0 + 16;
  float bias0 = bias ? bias[ncol0] : 0.f;
  float bias1 = bias ? bias[ncol1] : 0.f;
  int mbase = blockIdx.y * 32 + ((lane >= 16) ? 8 : 0);
#pragma unroll
  for (int r = 0; r < 8; ++r) {
    float v00 = acc00[r] + bias0, v01 = acc01[r] + bias1;
    float v10 = acc10[r] + bias0, v11 = acc11[r] + bias1;
    if (do_tanh) { v00 = tanhf(v00); v01 = tanhf(v01); v10 = tanhf(v10); v11 = tanhf(v11); }
    long m0 = mbase + r, m1 = mbase + 16 + r;
    out[m0 * out_stride + ncol0] = v00;
    out[m0 * out_stride + ncol1] = v01;
    out[m1 * out_stride + ncol0] = v10;
    out[m1 * out_stride + ncol1] = v11;
  }
}

// ---- Per-step fused LSTM cell + attention + A-pack. One WG per batch row.
__global__ __launch_bounds__(256) void lstm_attn_kernel(
    int t, const int* __restrict__ target, const float* __restrict__ emb,
    const float* __restrict__ W_x, const float* __restrict__ W_h,
    const float* __restrict__ b_lstm, const float* __restrict__ enc_proj,
    const float* __restrict__ enc_out, float* __restrict__ hS, float* __restrict__ cS,
    bf16* __restrict__ Afc) {
  int b = blockIdx.x, tid = threadIdx.x;
  __shared__ float xs[E_];
  __shared__ float hs[U_];
  __shared__ float zb[4 * U_];
  __shared__ float hn[U_];
  __shared__ float cx[U_];
  __shared__ float sc[256];
  __shared__ float sc2[L_];
  __shared__ float attn[L_];
  __shared__ float invsum;

  int tok = target[b * S_ + t];
  xs[tid] = emb[(long)tok * E_ + tid];  // E_ == 256 == blockDim
  for (int u = tid; u < U_; u += 256) hs[u] = hS[b * U_ + u];
  __syncthreads();

  // z = x @ W_x + h @ W_h + b   (each thread: 8 of 2048 outputs; coalesced W columns)
  for (int jj = 0; jj < 8; ++jj) {
    int u = tid + 256 * jj;
    float acc = b_lstm[u];
    const float* wx = W_x + u;
    for (int e = 0; e < E_; ++e) acc += xs[e] * wx[(long)e * 2048];
    const float* wh = W_h + u;
    for (int k = 0; k < U_; ++k) acc += hs[k] * wh[(long)k * 2048];
    zb[u] = acc;
  }
  __syncthreads();
  for (int u = tid; u < U_; u += 256) {
    float ig = 1.f / (1.f + __expf(-zb[u]));
    float fg = 1.f / (1.f + __expf(-zb[512 + u]));
    float gg = tanhf(zb[1024 + u]);
    float og = 1.f / (1.f + __expf(-zb[1536 + u]));
    float c = fg * cS[b * U_ + u] + ig * gg;
    cS[b * U_ + u] = c;
    float h = og * tanhf(c);
    hn[u] = h;
    hS[b * U_ + u] = h;
  }
  __syncthreads();

  // scores: 4 threads per l over U=512
  {
    int l = tid >> 2, part = tid & 3;
    const float* ep = enc_proj + ((long)b * L_ + l) * U_ + part * 128;
    float s = 0.f;
    for (int u = 0; u < 128; ++u) s += hn[part * 128 + u] * ep[u];
    sc[tid] = s;
  }
  __syncthreads();
  if (tid < L_) sc2[tid] = sc[4 * tid] + sc[4 * tid + 1] + sc[4 * tid + 2] + sc[4 * tid + 3];
  __syncthreads();
  if (tid == 0) {
    float m = sc2[0];
    for (int l = 1; l < L_; ++l) m = fmaxf(m, sc2[l]);
    float ssum = 0.f;
    for (int l = 0; l < L_; ++l) { float e = __expf(sc2[l] - m); attn[l] = e; ssum += e; }
    invsum = 1.f / ssum;
  }
  __syncthreads();
  if (tid < L_) attn[tid] *= invsum;
  __syncthreads();

  // ctx = attn @ encoder_outputs
  for (int u = tid; u < U_; u += 256) {
    float s = 0.f;
    const float* eo = enc_out + (long)b * L_ * U_ + u;
    for (int l = 0; l < L_; ++l) s += attn[l] * eo[(long)l * U_];
    cx[u] = s;
  }
  __syncthreads();

  // Pack row b of A = [ctx | h] (K=1024) into WMMA-A bf16 tiles ([mb][kb] layout)
  int mb = b >> 4;
  for (int idx = tid; idx < 1024; idx += 256) {
    int kb = idx >> 5;
    int r = idx & 31;
    int lane = (b & 15) + ((r >> 4) << 4);
    int j = r & 15;
    int k = kb * 32 + a_k_of(lane, j);
    float v = (k < 512) ? cx[k] : hn[k - 512];
    Afc[((long)mb * 32 + kb) * 512 + lane * 16 + j] = (bf16)v;
  }
}

// ---- In-place log_softmax over V for step t. One WG per batch row.
__global__ __launch_bounds__(256) void logsoftmax_kernel(float* __restrict__ out, int t) {
  int b = blockIdx.x, tid = threadIdx.x;
  float* row = out + ((long)b * T_ + t) * V_;
  __shared__ float red[256];
  float m = -1e30f;
  for (int v = tid; v < V_; v += 256) m = fmaxf(m, row[v]);
  red[tid] = m; __syncthreads();
  for (int s = 128; s > 0; s >>= 1) { if (tid < s) red[tid] = fmaxf(red[tid], red[tid + s]); __syncthreads(); }
  m = red[0]; __syncthreads();
  float ss = 0.f;
  for (int v = tid; v < V_; v += 256) ss += __expf(row[v] - m);
  red[tid] = ss; __syncthreads();
  for (int s = 128; s > 0; s >>= 1) { if (tid < s) red[tid] += red[tid + s]; __syncthreads(); }
  float lse = m + logf(red[0]);
  for (int v = tid; v < V_; v += 256) row[v] -= lse;
}

__global__ void init_state_kernel(const float* __restrict__ h0, const float* __restrict__ c0,
                                  float* __restrict__ hS, float* __restrict__ cS) {
  int i = blockIdx.x * 256 + threadIdx.x;
  if (i < B_ * U_) { hS[i] = h0[i]; cS[i] = c0[i]; }
}

extern "C" void kernel_launch(void* const* d_in, const int* in_sizes, int n_in,
                              void* d_out, int out_size, void* d_ws, size_t ws_size,
                              hipStream_t stream) {
  const int*   target = (const int*)d_in[0];
  const float* enc    = (const float*)d_in[1];
  const float* h0     = (const float*)d_in[2];
  const float* c0     = (const float*)d_in[3];
  const float* emb    = (const float*)d_in[4];
  const float* W_x    = (const float*)d_in[5];
  const float* W_h    = (const float*)d_in[6];
  const float* b_l    = (const float*)d_in[7];
  const float* Wa     = (const float*)d_in[8];
  const float* W_fc   = (const float*)d_in[9];
  const float* b_fc   = (const float*)d_in[10];
  float* out = (float*)d_out;

  // Workspace layout (~70 MB)
  char* w = (char*)d_ws;
  float* hS      = (float*)(w);                                   // 64 KB
  float* cS      = (float*)(w + (1u << 16));                      // 64 KB
  bf16*  Afc     = (bf16*) (w + 2u * (1u << 16));                 // 64 KB packed A (32x1024 bf16)
  float* encProj = (float*)(w + 3u * (1u << 16));                 // 4 MB
  bf16*  encA    = (bf16*) (w + 3u * (1u << 16) + 4194304u);      // 2 MB packed enc A
  bf16*  WaPk    = (bf16*) (w + 3u * (1u << 16) + 4194304u + 2097152u);            // 512 KB
  bf16*  WfcPk   = (bf16*) (w + 3u * (1u << 16) + 4194304u + 2097152u + 524288u);  // 64 MB

  init_state_kernel<<<64, 256, 0, stream>>>(h0, c0, hS, cS);

  // One-time packing (bf16, WMMA-native tile layouts)
  pack_b_kernel<<<64, 256, 0, stream>>>(Wa, WaPk, U_, U_);            // 512x512
  pack_b_kernel<<<8000, 256, 0, stream>>>(W_fc, WfcPk, 2 * U_, V_);   // 1024x32000
  pack_a_kernel<<<256, 256, 0, stream>>>(enc, encA, B_ * L_, U_);     // 2048x512

  // enc_proj = encoder_outputs @ Wa  (M=2048, N=512, K=512)
  gemm_bf16_wmma<<<dim3(2, 64), 256, 0, stream>>>(encA, WaPk, encProj, nullptr,
                                                  U_, (long)U_, 0);

  for (int t = 0; t < T_; ++t) {
    lstm_attn_kernel<<<B_, 256, 0, stream>>>(t, target, emb, W_x, W_h, b_l,
                                             encProj, enc, hS, cS, Afc);
    // logits_tanh = tanh([ctx|h] @ W_fc + b_fc) -> d_out[:, t, :]  (M=32, N=32000, K=1024)
    gemm_bf16_wmma<<<dim3(125, 1), 256, 0, stream>>>(Afc, WfcPk, out + (long)t * V_,
                                                     b_fc, 2 * U_, (long)T_ * V_, 1);
    logsoftmax_kernel<<<B_, 256, 0, stream>>>(out, t);
  }
}
